// KANConvolutionalLayer_4483945857426
// MI455X (gfx1250) — matrix operand order repaired and checked by reading, not compiled, not run on hardware
//
#include <hip/hip_runtime.h>

typedef __attribute__((ext_vector_type(16))) _Float16 v16h;
typedef __attribute__((ext_vector_type(8)))  float    v8f;

#define CIN   16
#define COUT  32
#define B_    16
#define H_    64
#define W_    64
#define NB    8
#define SLOTS 10            // 9 real features + 1 zero pad per Cin
#define KDIM  160           // CIN * SLOTS, = 5 chunks of 32
#define KCH   5
#define HP    66            // H + 2 halo
#define WP    66
#define NFRAG (9*KCH*2)     // taps * kchunks * ntiles = 90
#define FRAG_DW 256         // 32 lanes * 8 dwords
#define FEAT_OFFSET (128*1024)            // feature buffer offset in d_ws

#define CONV_THREADS 512    // 16 waves per workgroup -> stage weights 4x less
#define CONV_WAVES   (CONV_THREADS/32)

// ---------------------------------------------------------------------------
// Kernel 1: fold base_w + spline_w*scaler into WMMA B-fragments.
// Fragment f = tap*10 + q*2 + j holds B-tile (K = q*32..q*32+31, N = j*16..+15)
// in per-lane order: lane l<16 -> col N=j*16+l, K = q*32 + {2v,2v+1};
//                    lane l>=16 -> col N=j*16+(l-16), K = q*32+16 + {2v,2v+1}.
// Stored lane-major: dword[frag*256 + lane*8 + v] = (f16 Keven, f16 Kodd).
// ---------------------------------------------------------------------------
__global__ void kan_weight_prep(const float* __restrict__ bw,
                                const float* __restrict__ sw,
                                const float* __restrict__ sc,
                                unsigned int* __restrict__ wfrag)
{
    int id = blockIdx.x * blockDim.x + threadIdx.x;
    if (id >= NFRAG * 32) return;
    int lane = id & 31;
    int frag = id >> 5;
    int j   = frag & 1;
    int q   = (frag >> 1) % KCH;
    int tap = frag / (KCH * 2);
    int n   = j * 16 + (lane & 15);
    int koff = q * 32 + ((lane < 16) ? 0 : 16);

    unsigned int* dst = wfrag + frag * FRAG_DW + lane * 8;
#pragma unroll
    for (int v = 0; v < 8; v++) {
        _Float16 hv[2];
#pragma unroll
        for (int e = 0; e < 2; e++) {
            int kg = koff + 2 * v + e;        // global K slot in [0,160)
            int i  = kg / SLOTS;              // input channel
            int f  = kg % SLOTS;              // feature id
            float wv = 0.0f;
            int unit = (n * CIN + i) * 9 + tap;   // KANLinear index (o*CIN+i), tap
            if (f == 0)       wv = bw[unit];
            else if (f <= NB) wv = sw[unit * NB + (f - 1)] * sc[unit];
            hv[e] = (_Float16)wv;
        }
        union { _Float16 h[2]; unsigned int u; } u;
        u.h[0] = hv[0]; u.h[1] = hv[1];
        dst[v] = u.u;
    }
}

// ---------------------------------------------------------------------------
// Kernel 2: featurize x into f16 feature maps on a (HP x WP) halo grid.
// Halo pixels use x = 0 (matches reference zero-padding: silu(0)=0 but the
// spline bases of 0 are NONZERO and must be kept).
// Layout: feat[((b*HP+hh)*WP+ww)*160 + i*10 + f], slot f=9 is zero padding.
// ---------------------------------------------------------------------------
__global__ void kan_featurize(const float* __restrict__ x,
                              _Float16* __restrict__ feat)
{
    int idx = blockIdx.x * blockDim.x + threadIdx.x;
    if (idx >= B_ * CIN * HP * WP) return;
    int ww = idx % WP;
    int t  = idx / WP;
    int hh = t % HP;  t /= HP;
    int i  = t % CIN;
    int b  = t / CIN;

    float v = 0.0f;
    if (hh >= 1 && hh <= H_ && ww >= 1 && ww <= W_)
        v = x[((b * CIN + i) * H_ + (hh - 1)) * W_ + (ww - 1)];

    // base branch: silu
    float s = v * (1.0f / (1.0f + __expf(-v)));

    // Cox-de Boor, uniform knots t_j = 0.4*j - 2.2, j = 0..11
    float b0[11], b1[10], b2[9], b3[8];
#pragma unroll
    for (int j = 0; j < 11; j++) {
        float tj = 0.4f * j - 2.2f;
        b0[j] = (v >= tj && v < tj + 0.4f) ? 1.0f : 0.0f;
    }
#pragma unroll
    for (int j = 0; j < 10; j++) {
        float tj = 0.4f * j - 2.2f;
        b1[j] = (v - tj) * 2.5f * b0[j] + ((tj + 0.8f) - v) * 2.5f * b0[j + 1];
    }
#pragma unroll
    for (int j = 0; j < 9; j++) {
        float tj = 0.4f * j - 2.2f;
        b2[j] = (v - tj) * 1.25f * b1[j] + ((tj + 1.2f) - v) * 1.25f * b1[j + 1];
    }
#pragma unroll
    for (int j = 0; j < 8; j++) {
        float tj = 0.4f * j - 2.2f;
        b3[j] = (v - tj) * 0.8333333433f * b2[j] +
                ((tj + 1.6f) - v) * 0.8333333433f * b2[j + 1];
    }

    _Float16 f[SLOTS];
    f[0] = (_Float16)s;
#pragma unroll
    for (int j = 0; j < NB; j++) f[1 + j] = (_Float16)b3[j];
    f[9] = (_Float16)0.0f;

    int pix = (b * HP + hh) * WP + ww;
    unsigned int* dst =
        (unsigned int*)(feat + (size_t)pix * KDIM + i * SLOTS);
#pragma unroll
    for (int p = 0; p < 5; p++) {
        union { _Float16 h[2]; unsigned int u; } u;
        u.h[0] = f[2 * p]; u.h[1] = f[2 * p + 1];
        dst[p] = u.u;
    }
}

// ---------------------------------------------------------------------------
// Kernel 3: 3x3 conv as WMMA GEMM. One wave = 16 pixels (along W) x 32 chans.
// 90 x v_wmma_f32_16x16x32_f16 per wave. Weights staged in LDS (90 KB),
// shared by 16 waves per workgroup to amortize staging traffic.
// ---------------------------------------------------------------------------
__global__ __launch_bounds__(CONV_THREADS)
void kan_conv_wmma(const _Float16* __restrict__ feat,
                   const unsigned int* __restrict__ wfrag,
                   float* __restrict__ out)
{
    extern __shared__ unsigned int ldsW[];   // NFRAG*FRAG_DW dwords = 92160 B

    const int tid = threadIdx.x;
    // Cooperative stage of all weight fragments into LDS.
    {
        const uint4* src = (const uint4*)wfrag;
        uint4* dst = (uint4*)ldsW;
        const int n16 = NFRAG * FRAG_DW / 4;   // 5760 uint4
#pragma unroll 1
        for (int p = tid; p < n16; p += CONV_THREADS) dst[p] = src[p];
    }
    __syncthreads();

    const int wave = tid >> 5;
    const int lane = tid & 31;
    const int mtile = blockIdx.x * CONV_WAVES + wave;   // [0, 4096)
    const int wt = mtile & 3;                    // 16-pixel tile along W
    const int h  = (mtile >> 2) & (H_ - 1);
    const int b  = mtile >> 8;

    const int m      = lane & 15;                // pixel row of A-fragment
    const int hiHalf = lane >> 4;                // lane group -> K phase

    v8f c0 = {};   // channels 0..15
    v8f c1 = {};   // channels 16..31

#pragma unroll
    for (int ty = 0; ty < 3; ty++) {
#pragma unroll
        for (int tx = 0; tx < 3; tx++) {
            const int tap = ty * 3 + tx;
            const int pix = (b * HP + (h + ty)) * WP + (wt * 16 + m + tx);
            const char* ap = (const char*)feat +
                             (size_t)pix * (KDIM * 2) + hiHalf * 32;
#pragma unroll
            for (int q = 0; q < KCH; q++) {
                union { uint4 i4[2]; v16h h; } A, B0, B1;
                A.i4[0] = *(const uint4*)(ap + q * 64);
                A.i4[1] = *(const uint4*)(ap + q * 64 + 16);
                const int fb = (tap * 10 + q * 2) * FRAG_DW + lane * 8;
                B0.i4[0] = *(const uint4*)&ldsW[fb];
                B0.i4[1] = *(const uint4*)&ldsW[fb + 4];
                B1.i4[0] = *(const uint4*)&ldsW[fb + FRAG_DW];
                B1.i4[1] = *(const uint4*)&ldsW[fb + FRAG_DW + 4];
                c0 = __builtin_amdgcn_wmma_f32_16x16x32_f16(
                         false, A.h, false, B0.h, (short)0, c0, false, false);
                c1 = __builtin_amdgcn_wmma_f32_16x16x32_f16(
                         false, A.h, false, B1.h, (short)0, c1, false, false);
            }
        }
    }

    // D layout: lane -> N = lane&15 (+16 for c1), VGPR r -> pixel m = hiHalf*8 + r.
    const int n0 = lane & 15;
    const int wbase = wt * 16 + hiHalf * 8;
    union { v8f v; float4 f4[2]; } r0, r1;
    r0.v = c0; r1.v = c1;
    float* o0 = out + (((size_t)b * COUT + n0) * H_ + h) * W_ + wbase;
    float* o1 = out + (((size_t)b * COUT + 16 + n0) * H_ + h) * W_ + wbase;
    *(float4*)(o0)     = r0.f4[0];
    *(float4*)(o0 + 4) = r0.f4[1];
    *(float4*)(o1)     = r1.f4[0];
    *(float4*)(o1 + 4) = r1.f4[1];
}

// ---------------------------------------------------------------------------
extern "C" void kernel_launch(void* const* d_in, const int* in_sizes, int n_in,
                              void* d_out, int out_size, void* d_ws, size_t ws_size,
                              hipStream_t stream)
{
    const float* x        = (const float*)d_in[0];   // [16,16,64,64]
    const float* base_w   = (const float*)d_in[1];   // [512,9]
    const float* spline_w = (const float*)d_in[2];   // [512,9,8]
    const float* scaler   = (const float*)d_in[3];   // [512,9]
    float* out = (float*)d_out;                      // [16,32,64,64]

    unsigned int* wfrag = (unsigned int*)d_ws;
    _Float16* feat = (_Float16*)((char*)d_ws + FEAT_OFFSET);

    kan_weight_prep<<<(NFRAG * 32 + 127) / 128, 128, 0, stream>>>(
        base_w, spline_w, scaler, wfrag);

    const int nfeat = B_ * CIN * HP * WP;
    kan_featurize<<<(nfeat + 255) / 256, 256, 0, stream>>>(x, feat);

    const int nmtiles = B_ * H_ * (W_ / 16);         // 4096 waves total
    kan_conv_wmma<<<nmtiles / CONV_WAVES, CONV_THREADS,
                    NFRAG * FRAG_DW * 4, stream>>>(feat, wfrag, out);
}
